// ProbeMessagePassingLayer_16320875725329
// MI455X (gfx1250) — matrix operand order repaired and checked by reading, not compiled, not run on hardware
//
#include <hip/hip_runtime.h>
#include <hip/hip_bf16.h>
#include <stdint.h>

#define N_NODES   50000
#define N_EDGESC  640000
#define NODE_DIM  128
#define EDGE_DIM  128
#define HIDDEN    256
#define EIN_DIM   384   // 2*NODE_DIM + EDGE_DIM
#define NIN_DIM   256   // NODE_DIM + EDGE_DIM

typedef __attribute__((ext_vector_type(16))) __bf16 v16bf;
typedef __attribute__((ext_vector_type(8)))  float  v8f;

union BFrag { v16bf v; uint4 q[2]; };

__device__ __forceinline__ unsigned short f2bf(float f) {
  union { float f; unsigned u; } v; v.f = f;
  unsigned r = v.u + 0x7FFFu + ((v.u >> 16) & 1u);   // round-to-nearest-even
  return (unsigned short)(r >> 16);
}
__device__ __forceinline__ unsigned pack2(float a, float b) {
  return (unsigned)f2bf(a) | ((unsigned)f2bf(b) << 16);
}
__device__ __forceinline__ float gelu_tanh(float x) {
  float x3 = x * x * x;
  return 0.5f * x * (1.0f + tanhf(0.7978845608028654f * (x + 0.044715f * x3)));
}

// A-matrix 16x32 bf16 fragment from LDS (contiguous k order, same as the B packing).
// lane 0-15: M=lane, K = k0 + j; lane 16-31: same +16 on K.
__device__ __forceinline__ v16bf load_a_frag(const unsigned short* base, int lane,
                                             int stride, int k0) {
  int m    = lane & 15;
  int off8 = (lane >> 4) << 4;
  const unsigned short* p = base + m * stride + k0 + off8;
  BFrag u;
  u.q[0] = *(const uint4*)p;          // j=0..7   -> K = k0+off8 .. +7
  u.q[1] = *(const uint4*)(p + 8);    // j=8..15  -> K = k0+off8+8 .. +15
  return u.v;
}

// B-matrix 32x16 fragment, pre-swizzled in workspace: [frag][lane][16 halves]
__device__ __forceinline__ v16bf load_b_frag(const unsigned short* ws, int fragIdx, int lane) {
  const uint4* p = (const uint4*)(ws + ((size_t)(fragIdx * 32 + lane) << 4));
  BFrag u;
  u.q[0] = p[0];
  u.q[1] = p[1];
  return u.v;
}

// ---------------------------------------------------------------- prep kernels
__global__ void zero_f32_kernel(float* p, int n) {
  int i = blockIdx.x * blockDim.x + threadIdx.x;
  int stride = gridDim.x * blockDim.x;
  for (; i < n; i += stride) p[i] = 0.0f;
}

// Convert W[K][N] f32 -> bf16 B-fragment order: K = 32*kt + 16*(lane>=16) + j,
// N = 16*nt + (lane&15); flat dst index = (((kt*NT + nt)*32) + lane)*16 + j.
__global__ void cvt_weight_kernel(const float* __restrict__ src,
                                  unsigned short* __restrict__ dst,
                                  int Kdim, int Ndim) {
  int total = Kdim * Ndim;
  int NT = Ndim >> 4;
  int i = blockIdx.x * blockDim.x + threadIdx.x;
  int stride = gridDim.x * blockDim.x;
  for (; i < total; i += stride) {
    int j    = i & 15;
    int lane = (i >> 4) & 31;
    int t    = i >> 9;
    int nt   = t % NT;
    int kt   = t / NT;
    int K = kt * 32 + ((lane >> 4) << 4) + j;
    int N = nt * 16 + (lane & 15);
    dst[i] = f2bf(src[(size_t)K * Ndim + N]);
  }
}

// ---------------------------------------------------------------- edge kernel
#define EK_WAVES   4
#define EK_TILE    32                    // edges per wave (two 16-row WMMA tiles)
#define E_IN_STR   (EIN_DIM + 8)         // 392 halves: +8 pad -> banks 4*i
#define E_HID_STR  (HIDDEN + 8)          // 264 halves

__global__ void __launch_bounds__(128, 1)
edge_mlp_kernel(const float* __restrict__ nodes, const float* __restrict__ edges,
                const int* __restrict__ senders, const int* __restrict__ receivers,
                const unsigned short* __restrict__ w1, const float* __restrict__ b1,
                const unsigned short* __restrict__ w2, const float* __restrict__ b2,
                float* __restrict__ edges_out, float* __restrict__ agg) {
  extern __shared__ unsigned char smem[];
  const int tid  = threadIdx.x;
  const int lane = tid & 31;
  const int w    = tid >> 5;
  unsigned short* inb  = (unsigned short*)smem + (size_t)w * EK_TILE * E_IN_STR;
  unsigned short* hidb = (unsigned short*)smem
                       + (size_t)EK_WAVES * EK_TILE * E_IN_STR
                       + (size_t)w * EK_TILE * E_HID_STR;

  const int e0 = blockIdx.x * (EK_WAVES * EK_TILE) + w * EK_TILE;

  // per-lane edge indices (lane == local edge row)
  int s_l = senders[e0 + lane];
  int r_l = receivers[e0 + lane];

  // ---- gather + f32->bf16 into LDS: row = [sender(128) | receiver(128) | edge(128)]
  const float* eptr = edges + (size_t)e0 * EDGE_DIM;
  for (int m = 0; m < EK_TILE; ++m) {
    int sm = __shfl(s_l, m, 32);
    int rm = __shfl(r_l, m, 32);
    float4 a = *(const float4*)(nodes + (size_t)sm * NODE_DIM + lane * 4);
    float4 b = *(const float4*)(nodes + (size_t)rm * NODE_DIM + lane * 4);
    float4 c = *(const float4*)(eptr + (size_t)m * EDGE_DIM + lane * 4);
    unsigned short* row = inb + m * E_IN_STR;
    *(uint2*)(row +       lane * 4) = make_uint2(pack2(a.x, a.y), pack2(a.z, a.w));
    *(uint2*)(row + 128 + lane * 4) = make_uint2(pack2(b.x, b.y), pack2(b.z, b.w));
    *(uint2*)(row + 256 + lane * 4) = make_uint2(pack2(c.x, c.y), pack2(c.z, c.w));
  }

  // ---- GEMM1: [32x384] x [384x256] -> hidden, GELU, bf16 to LDS
  for (int nt = 0; nt < HIDDEN / 16; ++nt) {
    v8f acc0 = {}; v8f acc1 = {};
    for (int kt = 0; kt < EIN_DIM / 32; ++kt) {
      v16bf bf = load_b_frag(w1, kt * (HIDDEN / 16) + nt, lane);
      v16bf a0 = load_a_frag(inb,                    lane, E_IN_STR, kt * 32);
      v16bf a1 = load_a_frag(inb + 16 * E_IN_STR,    lane, E_IN_STR, kt * 32);
      acc0 = __builtin_amdgcn_wmma_f32_16x16x32_bf16(false, a0, false, bf, (short)0, acc0, false, false);
      acc1 = __builtin_amdgcn_wmma_f32_16x16x32_bf16(false, a1, false, bf, (short)0, acc1, false, false);
    }
    float bias = b1[nt * 16 + (lane & 15)];
    unsigned short* hcol = hidb + nt * 16 + (lane & 15);
    int mbase = (lane >> 4) << 3;
#pragma unroll
    for (int r = 0; r < 8; ++r) {
      hcol[(mbase + r)      * E_HID_STR] = f2bf(gelu_tanh(acc0[r] + bias));
      hcol[(mbase + r + 16) * E_HID_STR] = f2bf(gelu_tanh(acc1[r] + bias));
    }
  }

  // ---- GEMM2: [32x256] x [256x128] -> new_edges; residual store + scatter-add
  for (int nt = 0; nt < EDGE_DIM / 16; ++nt) {
    v8f acc0 = {}; v8f acc1 = {};
    for (int kt = 0; kt < HIDDEN / 32; ++kt) {
      v16bf bf = load_b_frag(w2, kt * (EDGE_DIM / 16) + nt, lane);
      v16bf a0 = load_a_frag(hidb,                   lane, E_HID_STR, kt * 32);
      v16bf a1 = load_a_frag(hidb + 16 * E_HID_STR,  lane, E_HID_STR, kt * 32);
      acc0 = __builtin_amdgcn_wmma_f32_16x16x32_bf16(false, a0, false, bf, (short)0, acc0, false, false);
      acc1 = __builtin_amdgcn_wmma_f32_16x16x32_bf16(false, a1, false, bf, (short)0, acc1, false, false);
    }
    int   n     = nt * 16 + (lane & 15);
    float bias  = b2[n];
    int   mbase = (lane >> 4) << 3;
#pragma unroll
    for (int r = 0; r < 8; ++r) {
      {
        int M = mbase + r;
        float val = acc0[r] + bias;
        size_t gi = (size_t)(e0 + M) * EDGE_DIM + n;
        edges_out[gi] = edges[gi] + val;
        int rv = __shfl(r_l, M, 32);
        atomicAdd(&agg[(size_t)rv * NODE_DIM + n], val);
      }
      {
        int M = mbase + r + 16;
        float val = acc1[r] + bias;
        size_t gi = (size_t)(e0 + M) * EDGE_DIM + n;
        edges_out[gi] = edges[gi] + val;
        int rv = __shfl(r_l, M, 32);
        atomicAdd(&agg[(size_t)rv * NODE_DIM + n], val);
      }
    }
  }
}

// ---------------------------------------------------------------- node kernel
#define NK_STR (NIN_DIM + 8)   // 264 halves, used for both input and hidden tiles

__global__ void __launch_bounds__(128, 1)
node_mlp_kernel(const float* __restrict__ nodes, const float* __restrict__ agg,
                const unsigned short* __restrict__ w1, const float* __restrict__ b1,
                const unsigned short* __restrict__ w2, const float* __restrict__ b2,
                float* __restrict__ nodes_out) {
  extern __shared__ unsigned char smem[];
  const int tid  = threadIdx.x;
  const int lane = tid & 31;
  const int w    = tid >> 5;
  unsigned short* inb  = (unsigned short*)smem + (size_t)w * EK_TILE * NK_STR;
  unsigned short* hidb = (unsigned short*)smem
                       + (size_t)EK_WAVES * EK_TILE * NK_STR
                       + (size_t)w * EK_TILE * NK_STR;

  const int n0 = blockIdx.x * (EK_WAVES * EK_TILE) + w * EK_TILE;

  // ---- gather: row = [node(128) | aggregated(128)] (clamped rows for the tail)
  for (int m = 0; m < EK_TILE; ++m) {
    int row  = n0 + m;
    int rowc = row < N_NODES ? row : (N_NODES - 1);
    float4 a = *(const float4*)(nodes + (size_t)rowc * NODE_DIM + lane * 4);
    float4 b = *(const float4*)(agg   + (size_t)rowc * NODE_DIM + lane * 4);
    unsigned short* r = inb + m * NK_STR;
    *(uint2*)(r +       lane * 4) = make_uint2(pack2(a.x, a.y), pack2(a.z, a.w));
    *(uint2*)(r + 128 + lane * 4) = make_uint2(pack2(b.x, b.y), pack2(b.z, b.w));
  }

  // ---- GEMM1: [32x256] x [256x256] -> hidden
  for (int nt = 0; nt < HIDDEN / 16; ++nt) {
    v8f acc0 = {}; v8f acc1 = {};
    for (int kt = 0; kt < NIN_DIM / 32; ++kt) {
      v16bf bf = load_b_frag(w1, kt * (HIDDEN / 16) + nt, lane);
      v16bf a0 = load_a_frag(inb,                 lane, NK_STR, kt * 32);
      v16bf a1 = load_a_frag(inb + 16 * NK_STR,   lane, NK_STR, kt * 32);
      acc0 = __builtin_amdgcn_wmma_f32_16x16x32_bf16(false, a0, false, bf, (short)0, acc0, false, false);
      acc1 = __builtin_amdgcn_wmma_f32_16x16x32_bf16(false, a1, false, bf, (short)0, acc1, false, false);
    }
    float bias = b1[nt * 16 + (lane & 15)];
    unsigned short* hcol = hidb + nt * 16 + (lane & 15);
    int mbase = (lane >> 4) << 3;
#pragma unroll
    for (int r = 0; r < 8; ++r) {
      hcol[(mbase + r)      * NK_STR] = f2bf(gelu_tanh(acc0[r] + bias));
      hcol[(mbase + r + 16) * NK_STR] = f2bf(gelu_tanh(acc1[r] + bias));
    }
  }

  // ---- GEMM2: [32x256] x [256x128] -> new_nodes; residual write (guarded)
  for (int nt = 0; nt < NODE_DIM / 16; ++nt) {
    v8f acc0 = {}; v8f acc1 = {};
    for (int kt = 0; kt < HIDDEN / 32; ++kt) {
      v16bf bf = load_b_frag(w2, kt * (NODE_DIM / 16) + nt, lane);
      v16bf a0 = load_a_frag(hidb,                lane, NK_STR, kt * 32);
      v16bf a1 = load_a_frag(hidb + 16 * NK_STR,  lane, NK_STR, kt * 32);
      acc0 = __builtin_amdgcn_wmma_f32_16x16x32_bf16(false, a0, false, bf, (short)0, acc0, false, false);
      acc1 = __builtin_amdgcn_wmma_f32_16x16x32_bf16(false, a1, false, bf, (short)0, acc1, false, false);
    }
    int   n     = nt * 16 + (lane & 15);
    float bias  = b2[n];
    int   mbase = (lane >> 4) << 3;
#pragma unroll
    for (int r = 0; r < 8; ++r) {
      {
        int row = n0 + mbase + r;
        if (row < N_NODES) {
          size_t gi = (size_t)row * NODE_DIM + n;
          nodes_out[gi] = nodes[gi] + (acc0[r] + bias);
        }
      }
      {
        int row = n0 + mbase + r + 16;
        if (row < N_NODES) {
          size_t gi = (size_t)row * NODE_DIM + n;
          nodes_out[gi] = nodes[gi] + (acc1[r] + bias);
        }
      }
    }
  }
}

// ---------------------------------------------------------------- launch
extern "C" void kernel_launch(void* const* d_in, const int* in_sizes, int n_in,
                              void* d_out, int out_size, void* d_ws, size_t ws_size,
                              hipStream_t stream) {
  const float* nodes     = (const float*)d_in[0];
  const float* edges     = (const float*)d_in[1];
  const int*   senders   = (const int*)d_in[2];
  const int*   receivers = (const int*)d_in[3];
  const float* ew1 = (const float*)d_in[4];
  const float* eb1 = (const float*)d_in[5];
  const float* ew2 = (const float*)d_in[6];
  const float* eb2 = (const float*)d_in[7];
  const float* nw1 = (const float*)d_in[8];
  const float* nb1 = (const float*)d_in[9];
  const float* nw2 = (const float*)d_in[10];
  const float* nb2 = (const float*)d_in[11];

  float* nodes_out = (float*)d_out;                           // also the scatter-add buffer
  float* edges_out = nodes_out + (size_t)N_NODES * NODE_DIM;

  unsigned short* w_ew1 = (unsigned short*)d_ws;              // 384x256 bf16 frags
  unsigned short* w_ew2 = w_ew1 + (size_t)EIN_DIM * HIDDEN;   // 256x128
  unsigned short* w_nw1 = w_ew2 + (size_t)HIDDEN * EDGE_DIM;  // 256x256
  unsigned short* w_nw2 = w_nw1 + (size_t)NIN_DIM * HIDDEN;   // 256x128

  // 1) zero aggregation buffer + bf16/fragment-swizzle all weights
  zero_f32_kernel<<<1024, 256, 0, stream>>>(nodes_out, N_NODES * NODE_DIM);
  cvt_weight_kernel<<<(EIN_DIM * HIDDEN + 255) / 256, 256, 0, stream>>>(ew1, w_ew1, EIN_DIM, HIDDEN);
  cvt_weight_kernel<<<(HIDDEN * EDGE_DIM + 255) / 256, 256, 0, stream>>>(ew2, w_ew2, HIDDEN, EDGE_DIM);
  cvt_weight_kernel<<<(NIN_DIM * HIDDEN + 255) / 256, 256, 0, stream>>>(nw1, w_nw1, NIN_DIM, HIDDEN);
  cvt_weight_kernel<<<(NIN_DIM * NODE_DIM + 255) / 256, 256, 0, stream>>>(nw2, w_nw2, NIN_DIM, NODE_DIM);

  // 2) edge MLP + residual + scatter-add (128 edges / block, exact: 640000/128)
  {
    size_t smemE = (size_t)EK_WAVES * EK_TILE * (E_IN_STR + E_HID_STR) * sizeof(unsigned short);
    edge_mlp_kernel<<<N_EDGESC / (EK_WAVES * EK_TILE), 128, smemE, stream>>>(
        nodes, edges, senders, receivers, w_ew1, eb1, w_ew2, eb2, edges_out, nodes_out);
  }

  // 3) node MLP + residual (reads aggregate from nodes_out, overwrites it)
  {
    size_t smemN = (size_t)EK_WAVES * EK_TILE * (2 * NK_STR) * sizeof(unsigned short);
    int grid = (N_NODES + EK_WAVES * EK_TILE - 1) / (EK_WAVES * EK_TILE);
    node_mlp_kernel<<<grid, 128, smemN, stream>>>(
        nodes, nodes_out, w_nw1, nb1, w_nw2, nb2, nodes_out);
  }
}
